// multi_head_Attention_1443109011569
// MI455X (gfx1250) — compile-verified, matrix operand order checked by reading, not run on hardware
//
#include <hip/hip_runtime.h>

// ---------------------------------------------------------------------------
// MI455X (gfx1250) multi-head attention, bf16 WMMA path, wave32.
// ---------------------------------------------------------------------------

typedef __attribute__((ext_vector_type(16))) __bf16 v16bf;
typedef __attribute__((ext_vector_type(8)))  float  v8f;
typedef __attribute__((ext_vector_type(4)))  unsigned int v4u;
typedef unsigned short bf16_t;

#define D_MODEL 1024
#define SEQ     2048
#define BATCH   2
#define NHEAD   16
#define DK      64
#define MROWS   (BATCH * SEQ)      // 4096
#define NQKV    (3 * D_MODEL)      // 3072

union Frag { v16bf v; v4u q[2]; };

static __device__ inline bf16_t f2bf(float f) {
    union { float f; unsigned int u; } c; c.f = f;
    unsigned int u = c.u;
    u = (u + 0x7FFFu + ((u >> 16) & 1u)) >> 16;   // round-to-nearest-even
    return (bf16_t)u;
}

static __device__ inline v8f vzero8() {
    v8f z;
#pragma unroll
    for (int i = 0; i < 8; ++i) z[i] = 0.0f;
    return z;
}

static __device__ inline v8f wmma_bf16(const Frag& a, const Frag& b, v8f c) {
    return __builtin_amdgcn_wmma_f32_16x16x32_bf16(
        false, a.v, false, b.v, (short)0, c, false, false);
}

// ---------------------------------------------------------------------------
// Prep kernels: f32 -> bf16 conversion + weight transposes.
// ---------------------------------------------------------------------------
__global__ void k_f32_to_bf16(const float* __restrict__ src, bf16_t* __restrict__ dst, int n) {
    int i = blockIdx.x * 256 + threadIdx.x;
    if (i < n) dst[i] = f2bf(src[i]);
}

// WqkvT[j][k] = W{q,k,v}[k][f], j in [0,3072), f = j%1024
__global__ void k_pack_wqkv_t(const float* __restrict__ Wq, const float* __restrict__ Wk,
                              const float* __restrict__ Wv, bf16_t* __restrict__ wt) {
    int i = blockIdx.x * 256 + threadIdx.x;          // 3072*1024 total
    int j = i >> 10, k = i & 1023;
    int seg = j >> 10, f = j & 1023;
    const float* W = (seg == 0) ? Wq : (seg == 1) ? Wk : Wv;
    wt[i] = f2bf(W[k * D_MODEL + f]);
}

__global__ void k_pack_wo_t(const float* __restrict__ Wo, bf16_t* __restrict__ wt) {
    int i = blockIdx.x * 256 + threadIdx.x;          // 1024*1024 total
    int j = i >> 10, k = i & 1023;
    wt[i] = f2bf(Wo[k * D_MODEL + j]);
}

// ---------------------------------------------------------------------------
// Register-tiled GEMM mainloop: one wave computes a 32x64 f32 tile.
// A row-major [.,lda] (k fast); Bt row-major [outcol][k] (i.e. B col-major).
// Fragment layout per ISA 7.12.2: lane l, base=l%16, klo=(l/16)*8,
// elements {klo..klo+7} u {klo+16..klo+23}  -> two contiguous 16B loads.
// ---------------------------------------------------------------------------
static __device__ inline void gemm_tile_32x64(const bf16_t* __restrict__ A,
                                              const bf16_t* __restrict__ Bt,
                                              int lda, int ldb, int kdim, v8f acc[8]) {
    const int lane = threadIdx.x & 31;
    const int r    = lane & 15;
    const int klo  = (lane >> 4) << 3;
    const bf16_t* a0 = A + r * lda + klo;
    const bf16_t* a1 = A + (r + 16) * lda + klo;
    const bf16_t* b0 = Bt + r * ldb + klo;
    for (int k = 0; k < kdim; k += 32) {
        Frag fa0, fa1;
        fa0.q[0] = *(const v4u*)(a0 + k);      fa0.q[1] = *(const v4u*)(a0 + k + 16);
        fa1.q[0] = *(const v4u*)(a1 + k);      fa1.q[1] = *(const v4u*)(a1 + k + 16);
#pragma unroll
        for (int ns = 0; ns < 4; ++ns) {
            Frag fb;
            const bf16_t* bp = b0 + ns * 16 * ldb + k;
            fb.q[0] = *(const v4u*)(bp);       fb.q[1] = *(const v4u*)(bp + 16);
            acc[ns]     = wmma_bf16(fa0, fb, acc[ns]);
            acc[4 + ns] = wmma_bf16(fa1, fb, acc[4 + ns]);
        }
    }
}

// ---------------------------------------------------------------------------
// Fused QKV projection GEMM: [4096,1024] x [1024,3072] + bias.
// Q scaled by 1/sqrt(dk)=0.125 (exact in bf16). Q,K: [B*H][S][64]; V: [B*H][64][S].
// Grid (NQKV/256, MROWS/64), block 256 (8 waves, 2x4 wave grid).
// ---------------------------------------------------------------------------
__global__ __launch_bounds__(256) void k_gemm_qkv(
    const bf16_t* __restrict__ xb, const bf16_t* __restrict__ wt,
    const float* __restrict__ bq, const float* __restrict__ bk, const float* __restrict__ bv,
    bf16_t* __restrict__ Qo, bf16_t* __restrict__ Ko, bf16_t* __restrict__ Vto) {
    const int wave = threadIdx.x >> 5, lane = threadIdx.x & 31;
    const int wm = wave >> 2, wn = wave & 3;
    const int rowbase = blockIdx.y * 64 + wm * 32;
    const int colbase = blockIdx.x * 256 + wn * 64;
    v8f acc[8];
#pragma unroll
    for (int i = 0; i < 8; ++i) acc[i] = vzero8();
    gemm_tile_32x64(xb + rowbase * D_MODEL, wt + colbase * D_MODEL, D_MODEL, D_MODEL, D_MODEL, acc);

    const int seg = colbase >> 10;              // 0=Q 1=K 2=V (tile never straddles)
    const float* bias = (seg == 0) ? bq : (seg == 1) ? bk : bv;
    const int hl = lane >> 4, n16 = lane & 15;
#pragma unroll
    for (int ms = 0; ms < 2; ++ms) {
#pragma unroll
        for (int ns = 0; ns < 4; ++ns) {
            const int col = colbase + ns * 16 + n16;
            const int f = col & 1023, h = f >> 6, fd = f & 63;
            const float bval = bias[f];
#pragma unroll
            for (int v = 0; v < 8; ++v) {
                const int row = rowbase + ms * 16 + v + hl * 8;
                const int b = row >> 11, sidx = row & 2047;
                const int bhn = b * NHEAD + h;
                const float val = acc[ms * 4 + ns][v] + bval;
                if (seg == 0)      Qo [(bhn * SEQ + sidx) * DK + fd] = f2bf(val * 0.125f);
                else if (seg == 1) Ko [(bhn * SEQ + sidx) * DK + fd] = f2bf(val);
                else               Vto[(bhn * DK + fd) * SEQ + sidx] = f2bf(val);
            }
        }
    }
}

// ---------------------------------------------------------------------------
// Flash attention: one wave per 16-query tile, 32 keys per iteration.
// Online softmax with 16-lane shuffle reductions; P transposed via LDS.
// Grid 512, block 256 (8 independent waves). LDS: 8 * 16 rows * stride 40.
// ---------------------------------------------------------------------------
__global__ __launch_bounds__(256) void k_attn(
    const bf16_t* __restrict__ Q, const bf16_t* __restrict__ Kc,
    const bf16_t* __restrict__ Vt, bf16_t* __restrict__ AO) {
    __shared__ bf16_t lds[8 * 16 * 40];
    const int wave = threadIdx.x >> 5, lane = threadIdx.x & 31;
    const int hl = lane >> 4, n16 = lane & 15, klo = hl * 8;
    const int tile = blockIdx.x * 8 + wave;          // 0..4095
    const int bh = tile >> 7, qt = tile & 127;
    const bf16_t* Qb = Q  + (bh * SEQ + qt * 16) * DK;
    const bf16_t* Kb = Kc + bh * SEQ * DK;
    const bf16_t* Vb = Vt + bh * DK * SEQ;
    bf16_t* ldsw = &lds[wave * 16 * 40];

    // Q fragments (dk=64 -> two K=32 chunks), resident for the whole pass
    Frag fq0, fq1;
    const bf16_t* qr = Qb + n16 * DK + klo;
    fq0.q[0] = *(const v4u*)(qr);        fq0.q[1] = *(const v4u*)(qr + 16);
    fq1.q[0] = *(const v4u*)(qr + 32);   fq1.q[1] = *(const v4u*)(qr + 48);

    v8f o[4];
#pragma unroll
    for (int i = 0; i < 4; ++i) o[i] = vzero8();
    v8f m, ssum;
#pragma unroll
    for (int v = 0; v < 8; ++v) { m[v] = -3.0e38f; ssum[v] = 0.0f; }

    for (int j = 0; j < SEQ; j += 32) {
        // ---- scores: S[16x32] = Q[16x64] * K[j..j+31]^T (scale folded into Q)
        v8f s0 = vzero8(), s1 = vzero8();
        const bf16_t* k0 = Kb + (j + n16) * DK + klo;
        const bf16_t* k1 = k0 + 16 * DK;
        Frag fk;
        fk.q[0] = *(const v4u*)(k0);      fk.q[1] = *(const v4u*)(k0 + 16);
        s0 = wmma_bf16(fq0, fk, s0);
        fk.q[0] = *(const v4u*)(k0 + 32); fk.q[1] = *(const v4u*)(k0 + 48);
        s0 = wmma_bf16(fq1, fk, s0);
        fk.q[0] = *(const v4u*)(k1);      fk.q[1] = *(const v4u*)(k1 + 16);
        s1 = wmma_bf16(fq0, fk, s1);
        fk.q[0] = *(const v4u*)(k1 + 32); fk.q[1] = *(const v4u*)(k1 + 48);
        s1 = wmma_bf16(fq1, fk, s1);

        // ---- online softmax (row = C-layout: vgpr v + 8*half; 16 cols per half-wave)
        v8f mnew, alpha, p0, p1, rsum;
#pragma unroll
        for (int v = 0; v < 8; ++v) {
            float t = fmaxf(s0[v], s1[v]);
            t = fmaxf(t, __shfl_xor(t, 1, 32));
            t = fmaxf(t, __shfl_xor(t, 2, 32));
            t = fmaxf(t, __shfl_xor(t, 4, 32));
            t = fmaxf(t, __shfl_xor(t, 8, 32));
            const float mn = fmaxf(m[v], t);
            mnew[v]  = mn;
            alpha[v] = __expf(m[v] - mn);
            const float e0 = __expf(s0[v] - mn);
            const float e1 = __expf(s1[v] - mn);
            p0[v] = e0; p1[v] = e1;
            float rs = e0 + e1;
            rs += __shfl_xor(rs, 1, 32);
            rs += __shfl_xor(rs, 2, 32);
            rs += __shfl_xor(rs, 4, 32);
            rs += __shfl_xor(rs, 8, 32);
            rsum[v] = rs;
        }
#pragma unroll
        for (int v = 0; v < 8; ++v) { ssum[v] = ssum[v] * alpha[v] + rsum[v]; m[v] = mnew[v]; }

        // ---- P (16x32) C-layout -> LDS -> A-fragment layout
#pragma unroll
        for (int v = 0; v < 8; ++v) {
            const int row = v + 8 * hl;
            ldsw[row * 40 + n16]      = f2bf(p0[v]);
            ldsw[row * 40 + 16 + n16] = f2bf(p1[v]);
        }
        asm volatile("s_wait_dscnt 0" ::: "memory");
        Frag fp;
        fp.q[0] = *(const v4u*)&ldsw[n16 * 40 + klo];
        fp.q[1] = *(const v4u*)&ldsw[n16 * 40 + klo + 16];

        // ---- O = O*alpha + P * V[j..j+31][:]   (Vt: contiguous along keys)
#pragma unroll
        for (int ns = 0; ns < 4; ++ns) {
#pragma unroll
            for (int v = 0; v < 8; ++v) o[ns][v] *= alpha[v];
            Frag fv;
            const bf16_t* vp = Vb + (ns * 16 + n16) * SEQ + j + klo;
            fv.q[0] = *(const v4u*)(vp); fv.q[1] = *(const v4u*)(vp + 16);
            o[ns] = wmma_bf16(fp, fv, o[ns]);
        }
    }

    // ---- epilogue: O/sum -> attn_out[B*S][H*dk] (bf16 for the output GEMM)
    const int b = bh >> 4, h = bh & 15;
#pragma unroll
    for (int ns = 0; ns < 4; ++ns) {
#pragma unroll
        for (int v = 0; v < 8; ++v) {
            const int sidx = qt * 16 + v + 8 * hl;
            const int grow = b * SEQ + sidx;
            const int col  = h * DK + ns * 16 + n16;
            AO[grow * D_MODEL + col] = f2bf(o[ns][v] / ssum[v]);
        }
    }
}

// ---------------------------------------------------------------------------
// Output projection: [4096,1024] x Wo + bo -> f32 d_out.
// ---------------------------------------------------------------------------
__global__ __launch_bounds__(256) void k_gemm_out(
    const bf16_t* __restrict__ ao, const bf16_t* __restrict__ wot,
    const float* __restrict__ bo, float* __restrict__ out) {
    const int wave = threadIdx.x >> 5, lane = threadIdx.x & 31;
    const int wm = wave >> 2, wn = wave & 3;
    const int rowbase = blockIdx.y * 64 + wm * 32;
    const int colbase = blockIdx.x * 256 + wn * 64;
    v8f acc[8];
#pragma unroll
    for (int i = 0; i < 8; ++i) acc[i] = vzero8();
    gemm_tile_32x64(ao + rowbase * D_MODEL, wot + colbase * D_MODEL, D_MODEL, D_MODEL, D_MODEL, acc);

    const int hl = lane >> 4, n16 = lane & 15;
#pragma unroll
    for (int ms = 0; ms < 2; ++ms) {
#pragma unroll
        for (int ns = 0; ns < 4; ++ns) {
            const int col = colbase + ns * 16 + n16;
            const float bval = bo[col];
#pragma unroll
            for (int v = 0; v < 8; ++v) {
                const int row = rowbase + ms * 16 + v + hl * 8;
                out[row * D_MODEL + col] = acc[ms * 4 + ns][v] + bval;
            }
        }
    }
}

// ---------------------------------------------------------------------------
extern "C" void kernel_launch(void* const* d_in, const int* in_sizes, int n_in,
                              void* d_out, int out_size, void* d_ws, size_t ws_size,
                              hipStream_t stream) {
    (void)in_sizes; (void)n_in; (void)out_size; (void)ws_size;
    const float* x  = (const float*)d_in[0];
    const float* Wq = (const float*)d_in[1];
    const float* bq = (const float*)d_in[2];
    const float* Wk = (const float*)d_in[3];
    const float* bk = (const float*)d_in[4];
    const float* Wv = (const float*)d_in[5];
    const float* bv = (const float*)d_in[6];
    const float* Wo = (const float*)d_in[7];
    const float* bo = (const float*)d_in[8];
    float* out = (float*)d_out;

    char* ws = (char*)d_ws;
    size_t off = 0;
    auto carve = [&](size_t bytes) { char* p = ws + off; off += (bytes + 255) & ~(size_t)255; return p; };
    bf16_t* x_bf  = (bf16_t*)carve((size_t)MROWS * D_MODEL * 2);   // 8 MB
    bf16_t* wqkvT = (bf16_t*)carve((size_t)NQKV  * D_MODEL * 2);   // 6 MB
    bf16_t* woT   = (bf16_t*)carve((size_t)D_MODEL * D_MODEL * 2); // 2 MB
    bf16_t* Qb    = (bf16_t*)carve((size_t)BATCH * NHEAD * SEQ * DK * 2); // 8 MB
    bf16_t* Kb    = (bf16_t*)carve((size_t)BATCH * NHEAD * SEQ * DK * 2); // 8 MB
    bf16_t* Vtb   = (bf16_t*)carve((size_t)BATCH * NHEAD * DK * SEQ * 2); // 8 MB
    bf16_t* AO    = (bf16_t*)carve((size_t)MROWS * D_MODEL * 2);   // 8 MB

    // 1) precision prep
    k_f32_to_bf16<<<(MROWS * D_MODEL) / 256, 256, 0, stream>>>(x, x_bf, MROWS * D_MODEL);
    k_pack_wqkv_t<<<(NQKV * D_MODEL) / 256, 256, 0, stream>>>(Wq, Wk, Wv, wqkvT);
    k_pack_wo_t<<<(D_MODEL * D_MODEL) / 256, 256, 0, stream>>>(Wo, woT);

    // 2) fused QKV projection
    k_gemm_qkv<<<dim3(NQKV / 256, MROWS / 64), 256, 0, stream>>>(
        x_bf, wqkvT, bq, bk, bv, Qb, Kb, Vtb);

    // 3) flash attention (4096 query tiles / 8 waves per block)
    k_attn<<<512, 256, 0, stream>>>(Qb, Kb, Vtb, AO);

    // 4) output projection
    k_gemm_out<<<dim3(D_MODEL / 256, MROWS / 64), 256, 0, stream>>>(AO, woT, bo, out);
}
